// SpatialRelPN_25864293056802
// MI455X (gfx1250) — compile-verified, matrix operand order checked by reading, not run on hardware
//
#include <hip/hip_runtime.h>
#include <cstdint>

typedef __attribute__((ext_vector_type(16))) __bf16          v16bf;
typedef __attribute__((ext_vector_type(16))) unsigned short  v16u;
typedef __attribute__((ext_vector_type(8)))  float           v8f;

#define B_IMGS   16
#define N_PROP   2048
#define D_FEAT   1024
#define H1       256
#define H2       64
#define PRE_NMS_ 256
#define POST_NMS_ 128
#define CANDS    4096   // 256 tiles/image * 16 candidates/tile

__device__ __forceinline__ unsigned short f2bf(float f) {
  unsigned int u = __float_as_uint(f);
  return (unsigned short)((u + 0x7FFFu + ((u >> 16) & 1u)) >> 16); // RNE
}

__device__ __forceinline__ v8f wmma_bf16(v16u a, v16u b, v8f c) {
  return __builtin_amdgcn_wmma_f32_16x16x32_bf16(
      false, __builtin_bit_cast(v16bf, a),
      false, __builtin_bit_cast(v16bf, b),
      (short)0, c, false, false);
}

// ---------------------------------------------------------------------------
// K0: convert f32 (K x N, row-major) -> bf16 transposed (N x K) so that WMMA
// B-fragments become contiguous 32B loads per lane.
// ---------------------------------------------------------------------------
__global__ void cvt_transpose(const float* __restrict__ src,
                              unsigned short* __restrict__ dst, int K, int N) {
  int i = blockIdx.x * blockDim.x + threadIdx.x;
  if (i >= K * N) return;
  int k = i / N, n = i % N;
  dst[(size_t)n * K + k] = f2bf(src[i]);
}

// ---------------------------------------------------------------------------
// K1: H = relu(F @ W1 + b1), both branches.
// Block = 32 rows x 256 cols, 8 waves in 2(row) x 4(col) layout.
// Each wave owns a 16x64 strip (4 accumulators): one A-fragment from LDS is
// reused by 4 back-to-back WMMAs -> load:WMMA issue ratio 5:4 instead of 2:1.
// A tile (32 x 1024 bf16 = 64 KB) staged once in LDS, shared by all 8 waves.
// ---------------------------------------------------------------------------
__global__ __launch_bounds__(256) void gemm1_relu(
    const float* __restrict__ feats, const unsigned short* __restrict__ w1t,
    const float* __restrict__ bs1, const float* __restrict__ bo1,
    unsigned short* __restrict__ Hout) {
  __shared__ unsigned short As[32 * D_FEAT]; // 64 KB bf16 A tile
  int z = blockIdx.z, b = z >> 1, br = z & 1;
  int rowBase = blockIdx.x * 32;
  const float* F = feats + ((size_t)b * N_PROP + rowBase) * D_FEAT;
  for (int i = threadIdx.x; i < 32 * D_FEAT; i += 256) As[i] = f2bf(F[i]);
  __syncthreads();

  int lane = threadIdx.x & 31, wave = threadIdx.x >> 5;
  int rowHalf = wave & 1;           // 0/1 -> which 16-row half
  int colBase = (wave >> 1) * 64;   // 4 col groups of 64
  int khalf = (lane >> 4) << 4;
  const unsigned short* arow = &As[(rowHalf * 16 + (lane & 15)) * D_FEAT];
  const unsigned short* wbase = w1t + (size_t)br * H1 * D_FEAT;
  const unsigned short* wcol0 = wbase + (size_t)(colBase +  0 + (lane & 15)) * D_FEAT;
  const unsigned short* wcol1 = wbase + (size_t)(colBase + 16 + (lane & 15)) * D_FEAT;
  const unsigned short* wcol2 = wbase + (size_t)(colBase + 32 + (lane & 15)) * D_FEAT;
  const unsigned short* wcol3 = wbase + (size_t)(colBase + 48 + (lane & 15)) * D_FEAT;

  v8f zero = {0.f, 0.f, 0.f, 0.f, 0.f, 0.f, 0.f, 0.f};
  v8f c0 = zero, c1 = zero, c2 = zero, c3 = zero;
#pragma unroll 2
  for (int k = 0; k < D_FEAT; k += 32) {
    // prefetch next k-chunk of the streamed B columns (global_prefetch_b8)
    __builtin_prefetch(wcol0 + k + 128, 0, 0);
    __builtin_prefetch(wcol2 + k + 128, 0, 0);
    v16u au = *(const v16u*)(arow + k + khalf);    // one LDS fragment ...
    v16u b0 = *(const v16u*)(wcol0 + k + khalf);
    v16u b1 = *(const v16u*)(wcol1 + k + khalf);
    v16u b2 = *(const v16u*)(wcol2 + k + khalf);
    v16u b3 = *(const v16u*)(wcol3 + k + khalf);
    c0 = wmma_bf16(au, b0, c0);                    // ... feeds 4 WMMAs
    c1 = wmma_bf16(au, b1, c1);
    c2 = wmma_bf16(au, b2, c2);
    c3 = wmma_bf16(au, b3, c3);
  }

  const float* bias = br ? bo1 : bs1;
  unsigned short* Hb = Hout + (size_t)br * B_IMGS * N_PROP * H1
                            + (size_t)b * N_PROP * H1;
  v8f acc[4] = {c0, c1, c2, c3};
#pragma unroll
  for (int t = 0; t < 4; ++t) {
    int col = colBase + t * 16 + (lane & 15);
    float bv = bias[col];
#pragma unroll
    for (int r = 0; r < 8; ++r) {
      int row = rowBase + rowHalf * 16 + r + ((lane >> 4) << 3);
      float v = acc[t][r] + bv;
      Hb[(size_t)row * H1 + col] = f2bf(v > 0.f ? v : 0.f);
    }
  }
}

// ---------------------------------------------------------------------------
// K2: subj/obj = H @ W2 + b2  (2048x256 @ 256x64).
// 4 waves per block; each wave owns a full 16x64 strip (4 accumulators),
// so each A-fragment load feeds 4 WMMAs. W2T (32 KB) lives in L2/L0.
// ---------------------------------------------------------------------------
__global__ __launch_bounds__(128) void gemm2(
    const unsigned short* __restrict__ Hin, const unsigned short* __restrict__ w2t,
    const float* __restrict__ bs2, const float* __restrict__ bo2,
    unsigned short* __restrict__ SO) {
  int z = blockIdx.y, b = z >> 1, br = z & 1;
  int lane = threadIdx.x & 31, wave = threadIdx.x >> 5;
  int rowBase = blockIdx.x * 64 + wave * 16;
  int khalf = (lane >> 4) << 4;
  const unsigned short* Hb = Hin + (size_t)br * B_IMGS * N_PROP * H1
                                 + (size_t)b * N_PROP * H1;
  const unsigned short* arow = Hb + (size_t)(rowBase + (lane & 15)) * H1;
  const unsigned short* wb = w2t + (size_t)br * H2 * H1;
  const unsigned short* wcol0 = wb + (size_t)( 0 + (lane & 15)) * H1;
  const unsigned short* wcol1 = wb + (size_t)(16 + (lane & 15)) * H1;
  const unsigned short* wcol2 = wb + (size_t)(32 + (lane & 15)) * H1;
  const unsigned short* wcol3 = wb + (size_t)(48 + (lane & 15)) * H1;

  v8f zero = {0.f, 0.f, 0.f, 0.f, 0.f, 0.f, 0.f, 0.f};
  v8f c0 = zero, c1 = zero, c2 = zero, c3 = zero;
#pragma unroll
  for (int k = 0; k < H1; k += 32) {
    v16u au = *(const v16u*)(arow + k + khalf);
    v16u b0 = *(const v16u*)(wcol0 + k + khalf);
    v16u b1 = *(const v16u*)(wcol1 + k + khalf);
    v16u b2 = *(const v16u*)(wcol2 + k + khalf);
    v16u b3 = *(const v16u*)(wcol3 + k + khalf);
    c0 = wmma_bf16(au, b0, c0);
    c1 = wmma_bf16(au, b1, c1);
    c2 = wmma_bf16(au, b2, c2);
    c3 = wmma_bf16(au, b3, c3);
  }

  const float* bias = br ? bo2 : bs2;
  unsigned short* O = SO + (size_t)br * B_IMGS * N_PROP * H2
                         + (size_t)b * N_PROP * H2;
  v8f acc[4] = {c0, c1, c2, c3};
#pragma unroll
  for (int t = 0; t < 4; ++t) {
    int col = t * 16 + (lane & 15);
    float bv = bias[col];
#pragma unroll
    for (int r = 0; r < 8; ++r) {
      int row = rowBase + r + ((lane >> 4) << 3);
      O[(size_t)row * H2 + col] = f2bf(acc[t][r] + bv);
    }
  }
}

// ---------------------------------------------------------------------------
// K3: 128x128 tile of sigmoid(subj @ obj^T), fused hierarchical top-k.
// Each wave: 16x128 strip = 8 accumulators, 2 WMMA k-steps each (K=64).
// Per-thread top-4 -> LDS(1024) -> 16 block argmax rounds -> 16 candidates.
// The 4M-score tensor never touches HBM.
// ---------------------------------------------------------------------------
__global__ __launch_bounds__(256) void score_topk(
    const unsigned short* __restrict__ SO,
    float* __restrict__ cand_s, int* __restrict__ cand_i) {
  __shared__ float ts[1024];
  __shared__ int   ti[1024];
  __shared__ float rs[256];
  __shared__ int   ri[256];
  int b = blockIdx.z;
  int lane = threadIdx.x & 31, wave = threadIdx.x >> 5;
  int rowBase = blockIdx.y * 128 + wave * 16;
  int colBase0 = blockIdx.x * 128;
  int khalf = (lane >> 4) << 4;
  const unsigned short* subj = SO + (size_t)b * N_PROP * H2;
  const unsigned short* obj  = SO + (size_t)B_IMGS * N_PROP * H2
                                  + (size_t)b * N_PROP * H2;
  const unsigned short* arow = subj + (size_t)(rowBase + (lane & 15)) * H2;

  v8f zero = {0.f, 0.f, 0.f, 0.f, 0.f, 0.f, 0.f, 0.f};
  v8f acc[8];
#pragma unroll
  for (int t = 0; t < 8; ++t) acc[t] = zero;

#pragma unroll
  for (int k = 0; k < H2; k += 32) {
    v16u au = *(const v16u*)(arow + k + khalf);
#pragma unroll
    for (int t = 0; t < 8; ++t) {
      const unsigned short* brow =
          obj + (size_t)(colBase0 + t * 16 + (lane & 15)) * H2;
      v16u bu = *(const v16u*)(brow + k + khalf);
      acc[t] = wmma_bf16(au, bu, acc[t]);
    }
  }

  // sigmoid + per-thread top-4 (sorted insertion)
  float v0 = -3.4e38f, v1 = -3.4e38f, v2 = -3.4e38f, v3 = -3.4e38f;
  int   j0 = 0, j1 = 0, j2 = 0, j3 = 0;
#pragma unroll
  for (int t = 0; t < 8; ++t) {
    int col = colBase0 + t * 16 + (lane & 15);
#pragma unroll
    for (int r = 0; r < 8; ++r) {
      int row = rowBase + r + ((lane >> 4) << 3);
      float sv = 1.f / (1.f + __expf(-acc[t][r]));
      int id = (row << 11) | col;  // flat index row*2048+col
      if (sv > v3) {
        if (sv > v0)      { v3=v2;j3=j2; v2=v1;j2=j1; v1=v0;j1=j0; v0=sv;j0=id; }
        else if (sv > v1) { v3=v2;j3=j2; v2=v1;j2=j1; v1=sv;j1=id; }
        else if (sv > v2) { v3=v2;j3=j2; v2=sv;j2=id; }
        else              { v3=sv;j3=id; }
      }
    }
  }
  int t4 = threadIdx.x * 4;
  ts[t4+0]=v0; ti[t4+0]=j0; ts[t4+1]=v1; ti[t4+1]=j1;
  ts[t4+2]=v2; ti[t4+2]=j2; ts[t4+3]=v3; ti[t4+3]=j3;
  __syncthreads();

  int blkLocal = blockIdx.y * 16 + blockIdx.x;
  float* cso = cand_s + (size_t)b * CANDS + blkLocal * 16;
  int*   cio = cand_i + (size_t)b * CANDS + blkLocal * 16;
  for (int round = 0; round < 16; ++round) {
    float mv = -3.4e38f; int me = 0;
    for (int e = threadIdx.x; e < 1024; e += 256)
      if (ts[e] > mv) { mv = ts[e]; me = e; }
    rs[threadIdx.x] = mv; ri[threadIdx.x] = me;
    __syncthreads();
    for (int s = 128; s > 0; s >>= 1) {
      if (threadIdx.x < s && rs[threadIdx.x + s] > rs[threadIdx.x]) {
        rs[threadIdx.x] = rs[threadIdx.x + s];
        ri[threadIdx.x] = ri[threadIdx.x + s];
      }
      __syncthreads();
    }
    if (threadIdx.x == 0) {
      int w = ri[0];
      cso[round] = rs[0];
      cio[round] = ti[w];
      ts[w] = -3.4e38f;
    }
    __syncthreads();
  }
}

// ---------------------------------------------------------------------------
// K4: per image — exact top-256 over 4096 candidates, inter boxes, greedy NMS,
// stable compaction, write (pairs, scores, inter, valid).
// ---------------------------------------------------------------------------
__global__ __launch_bounds__(256) void nms_finalize(
    const float* __restrict__ proposals,
    const float* __restrict__ cand_s, const int* __restrict__ cand_i,
    int* __restrict__ outPairs, float* __restrict__ outScores,
    float* __restrict__ outInter, float* __restrict__ outValid) {
  __shared__ float cs[CANDS];
  __shared__ int   ci[CANDS];
  __shared__ float rs[256];
  __shared__ int   ri[256];
  __shared__ float top_s[PRE_NMS_];
  __shared__ int   top_i[PRE_NMS_];
  __shared__ float ib[PRE_NMS_][4];
  __shared__ float area[PRE_NMS_];
  __shared__ int   keep[PRE_NMS_];
  __shared__ int   sel[POST_NMS_];
  __shared__ int   nsel;

  int b = blockIdx.x, tid = threadIdx.x;
  for (int i = tid; i < CANDS; i += 256) {
    cs[i] = cand_s[(size_t)b * CANDS + i];
    ci[i] = cand_i[(size_t)b * CANDS + i];
  }
  __syncthreads();

  for (int round = 0; round < PRE_NMS_; ++round) {
    float mv = -3.4e38f; int me = 0;
    for (int e = tid; e < CANDS; e += 256)
      if (cs[e] > mv) { mv = cs[e]; me = e; }
    rs[tid] = mv; ri[tid] = me;
    __syncthreads();
    for (int s = 128; s > 0; s >>= 1) {
      if (tid < s && rs[tid + s] > rs[tid]) { rs[tid]=rs[tid+s]; ri[tid]=ri[tid+s]; }
      __syncthreads();
    }
    if (tid == 0) { int w = ri[0]; top_s[round]=rs[0]; top_i[round]=ci[w]; cs[w]=-3.4e38f; }
    __syncthreads();
  }

  const float* P = proposals + (size_t)b * N_PROP * 4;
  if (tid < PRE_NMS_) {
    int id = top_i[tid], si = id >> 11, oi = id & 2047;
    float x1 = fmaxf(P[si*4+0], P[oi*4+0]);
    float y1 = fmaxf(P[si*4+1], P[oi*4+1]);
    float x2 = fminf(P[si*4+2], P[oi*4+2]);
    float y2 = fminf(P[si*4+3], P[oi*4+3]);
    ib[tid][0]=x1; ib[tid][1]=y1; ib[tid][2]=x2; ib[tid][3]=y2;
    area[tid] = (x2 - x1) * (y2 - y1);
    keep[tid] = 1;
  }
  __syncthreads();

  for (int i = 0; i < PRE_NMS_; ++i) {
    if (tid < PRE_NMS_ && tid > i && keep[i] && keep[tid]) {
      float xx1 = fmaxf(ib[i][0], ib[tid][0]);
      float yy1 = fmaxf(ib[i][1], ib[tid][1]);
      float xx2 = fminf(ib[i][2], ib[tid][2]);
      float yy2 = fminf(ib[i][3], ib[tid][3]);
      float w = fmaxf(xx2 - xx1, 0.f), h = fmaxf(yy2 - yy1, 0.f);
      float inter = w * h;
      float iou = inter / (area[i] + area[tid] - inter);
      if (iou > 0.7f) keep[tid] = 0;
    }
    __syncthreads();
  }

  if (tid == 0) {
    int c = 0;
    for (int t = 0; t < PRE_NMS_ && c < POST_NMS_; ++t) if (keep[t]) sel[c++] = t;
    nsel = c;
  }
  __syncthreads();

  if (tid < POST_NMS_) {
    size_t ob = (size_t)b * POST_NMS_;
    if (tid < nsel) {
      int s = sel[tid], id = top_i[s];
      outPairs[(ob + tid) * 2 + 0] = id >> 11;
      outPairs[(ob + tid) * 2 + 1] = id & 2047;
      outScores[ob + tid] = top_s[s];
      outInter[(ob + tid) * 4 + 0] = ib[s][0];
      outInter[(ob + tid) * 4 + 1] = ib[s][1];
      outInter[(ob + tid) * 4 + 2] = ib[s][2];
      outInter[(ob + tid) * 4 + 3] = ib[s][3];
      outValid[ob + tid] = 1.f;
    } else {
      outPairs[(ob + tid) * 2 + 0] = 0;
      outPairs[(ob + tid) * 2 + 1] = 0;
      outScores[ob + tid] = 0.f;
      outInter[(ob + tid) * 4 + 0] = 0.f;
      outInter[(ob + tid) * 4 + 1] = 0.f;
      outInter[(ob + tid) * 4 + 2] = 0.f;
      outInter[(ob + tid) * 4 + 3] = 0.f;
      outValid[ob + tid] = 0.f;
    }
  }
}

// ---------------------------------------------------------------------------
extern "C" void kernel_launch(void* const* d_in, const int* in_sizes, int n_in,
                              void* d_out, int out_size, void* d_ws, size_t ws_size,
                              hipStream_t stream) {
  (void)in_sizes; (void)n_in; (void)out_size; (void)ws_size;
  const float* feats     = (const float*)d_in[0];
  const float* proposals = (const float*)d_in[1];
  const float* Ws1 = (const float*)d_in[2];
  const float* bs1 = (const float*)d_in[3];
  const float* Ws2 = (const float*)d_in[4];
  const float* bs2 = (const float*)d_in[5];
  const float* Wo1 = (const float*)d_in[6];
  const float* bo1 = (const float*)d_in[7];
  const float* Wo2 = (const float*)d_in[8];
  const float* bo2 = (const float*)d_in[9];

  char* ws = (char*)d_ws;
  size_t o = 0;
  unsigned short* w1t = (unsigned short*)(ws + o); o += (size_t)2 * H1 * D_FEAT * 2;
  unsigned short* w2t = (unsigned short*)(ws + o); o += (size_t)2 * H2 * H1 * 2;
  unsigned short* Hbf = (unsigned short*)(ws + o); o += (size_t)2 * B_IMGS * N_PROP * H1 * 2;
  unsigned short* SO  = (unsigned short*)(ws + o); o += (size_t)2 * B_IMGS * N_PROP * H2 * 2;
  float* cand_s = (float*)(ws + o); o += (size_t)B_IMGS * CANDS * 4;
  int*   cand_i = (int*)(ws + o);   o += (size_t)B_IMGS * CANDS * 4;

  cvt_transpose<<<(D_FEAT * H1 + 255) / 256, 256, 0, stream>>>(Ws1, w1t, D_FEAT, H1);
  cvt_transpose<<<(D_FEAT * H1 + 255) / 256, 256, 0, stream>>>(Wo1, w1t + (size_t)H1 * D_FEAT, D_FEAT, H1);
  cvt_transpose<<<(H1 * H2 + 255) / 256, 256, 0, stream>>>(Ws2, w2t, H1, H2);
  cvt_transpose<<<(H1 * H2 + 255) / 256, 256, 0, stream>>>(Wo2, w2t + (size_t)H2 * H1, H1, H2);

  gemm1_relu<<<dim3(64, 1, 32), 256, 0, stream>>>(feats, w1t, bs1, bo1, Hbf);
  gemm2<<<dim3(32, 32), 128, 0, stream>>>(Hbf, w2t, bs2, bo2, SO);
  score_topk<<<dim3(16, 16, 16), 256, 0, stream>>>(SO, cand_s, cand_i);

  int*   outPairs  = (int*)d_out;
  float* outScores = (float*)d_out + (size_t)B_IMGS * POST_NMS_ * 2;
  float* outInter  = outScores + (size_t)B_IMGS * POST_NMS_;
  float* outValid  = outInter + (size_t)B_IMGS * POST_NMS_ * 4;
  nms_finalize<<<B_IMGS, 256, 0, stream>>>(proposals, cand_s, cand_i,
                                           outPairs, outScores, outInter, outValid);
}
